// MultiHeadAttention_35734127903180
// MI455X (gfx1250) — matrix operand order repaired and checked by reading, not compiled, not run on hardware
//
#include <hip/hip_runtime.h>
#include <hip/hip_bf16.h>

// ---------------------------------------------------------------------------
// MultiHeadAttention + BatchNorm + ReLU for MI455X (gfx1250, wave32, WMMA)
// All matmuls: v_wmma_f32_16x16x32_bf16 (fp32 accumulate).
// A-tile staging in the GEMM uses the Tensor Data Mover (tensor_load_to_lds,
// clang-23 6-arg builtin form), guarded; else cooperative loads.
//   B=16, L=1024, IN_DIM=HID=512, H=8, HD=64, M = B*L = 16384 rows.
// ---------------------------------------------------------------------------

typedef __attribute__((ext_vector_type(16))) __bf16 v16bf;
typedef __attribute__((ext_vector_type(8)))  float  v8f;
typedef __attribute__((ext_vector_type(4)))  unsigned int u32x4;
typedef __attribute__((ext_vector_type(8)))  int i32x8;
typedef __attribute__((ext_vector_type(4)))  int i32x4;

#define NB    16
#define NL    1024
#define DIN   512
#define NHID  512
#define NH    8
#define NHD   64
#define NM    (NB * NL)        // 16384 rows
#define EPSN  1e-5f

#if defined(__HIP_DEVICE_COMPILE__) &&                                   \
    __has_builtin(__builtin_amdgcn_tensor_load_to_lds) &&                \
    __has_builtin(__builtin_amdgcn_s_wait_tensorcnt)
#define USE_TDM 1
#else
#define USE_TDM 0
#endif

// ------------------------------ fp32 -> bf16 -------------------------------
__global__ void f32_to_bf16_kernel(const float* __restrict__ in,
                                   __bf16* __restrict__ out, int n) {
  int i = blockIdx.x * blockDim.x + threadIdx.x;
  if (i < n) out[i] = (__bf16)in[i];
}

// ------------------------------ WMMA GEMM ----------------------------------
// C[M,512] = A[M,512](bf16) * W[512,512](bf16 row-major = B^T) + bias
// Block: 256 threads (8 waves), tile 256(M) x 64(N), K stepped by 32.
// Wave w computes rows [w*32, w*32+32) x 64 cols: 8 f32 accumulators,
// 8 WMMAs per k-step against 2 A-fragments + 4 shared B-fragments.
//
// ISA fragment layouts (CDNA5 ISA):
//  A 16x32 bf16: lane<16 holds row m, halfs = K{0..7,16..23};
//                lane>=16 holds row m, halfs = K{8..15,24..31}  (2x16B loads)
//  B 32x16 bf16: lane holds column n=lane%16 of B (= row n of W),
//                halfs = K{g..g+7,16+g..16+g+7}, g=(lane/16)*8   (2x16B loads)
//  C/D 16x16 f32: vgpr r, lane group g: element (m=r+8g, n=lane%16)
template <bool OUT_F32>
__global__ void __launch_bounds__(256)
gemm_bf16_wmma(const __bf16* __restrict__ A, const __bf16* __restrict__ W,
               const float* __restrict__ bias, void* __restrict__ Cout) {
  __shared__ __bf16 As[256 * 32];   // 16 KB, one TDM tile
  __shared__ __bf16 Bs[64 * 32];    //  4 KB

  const int tid  = threadIdx.x;
  const int lane = tid & 31;
  const int wave = tid >> 5;
  const int ln   = lane & 15;
  const int hi   = lane >> 4;               // lane group 0/1
  const int m0   = blockIdx.x * 256;
  const int n0   = blockIdx.y * 64;

  v8f acc[2][4] = {};

  for (int k0 = 0; k0 < 512; k0 += 32) {
#if USE_TDM
    // --- Tensor Data Mover: stage A tile 256 rows x 32 cols (bf16) ---
    // D# per CDNA5 ISA:
    //   group0: count=1, lds_addr, global_addr, type=2
    //   group1: data_size=2B, tensor_dim0=512, tensor_dim1=NM,
    //           tile_dim0=32, tile_dim1=256, tensor_dim0_stride=512
    if (wave == 0) {
      unsigned long long ga =
          (unsigned long long)(const void*)(A + (size_t)m0 * 512 + k0);
      unsigned int lds = (unsigned int)(size_t)(void*)&As[0];
      u32x4 g0;
      g0[0] = 1u;                                   // count=1 (valid, user)
      g0[1] = lds;                                  // lds_addr
      g0[2] = (unsigned int)ga;                     // global_addr[31:0]
      g0[3] = (unsigned int)((ga >> 32) & 0x1FFFFFFu) | (2u << 30); // type=2
      i32x8 g1;
      g1[0] = (int)(1u << 16);      // data_size=1 (2 bytes/elem)
      g1[1] = (int)(512u << 16);    // tensor_dim0 = 512 (bits 79:48, lo16)
      g1[2] = (int)(((unsigned)NM & 0xFFFFu) << 16); // tensor_dim1 lo16 @95:80
      g1[3] = (int)(32u << 16);     // tensor_dim1 hi=0; tile_dim0=32 @127:112
      g1[4] = 256;                  // tile_dim1=256 @143:128; tile_dim2=0
      g1[5] = 512;                  // tensor_dim0_stride[31:0]
      g1[6] = 0;
      g1[7] = 0;
      i32x4 z4 = {0, 0, 0, 0};
      i32x8 z8 = {0, 0, 0, 0, 0, 0, 0, 0};
      // clang-23 / therock-10.0 6-arg form (probe: "expected 6" diagnostic)
      __builtin_amdgcn_tensor_load_to_lds(g0, g1, z4, z4, z8, 0);
      __builtin_amdgcn_s_wait_tensorcnt(0);
    }
#else
    // cooperative A tile: each thread copies one 32-elem row (64B)
    {
      int row = tid;
      const uint4* src = (const uint4*)(A + (size_t)(m0 + row) * 512 + k0);
      uint4* dst = (uint4*)(As + row * 32);
#pragma unroll
      for (int i = 0; i < 4; ++i) dst[i] = src[i];
      if (k0 + 32 < 512)  // -> global_prefetch_b8
        __builtin_prefetch(A + (size_t)(m0 + row) * 512 + k0 + 32, 0, 0);
    }
#endif
    // cooperative B tile: 64x32 bf16 (W rows n0..n0+63), 8 elems per thread
    {
      int row = tid >> 2;
      int kc  = (tid & 3) * 8;
      *(uint4*)(Bs + row * 32 + kc) =
          *(const uint4*)(W + (size_t)(n0 + row) * 512 + k0 + kc);
    }
    __syncthreads();

    union { v16bf v; uint4 q[2]; } af[2];
#pragma unroll
    for (int s = 0; s < 2; ++s) {
      const __bf16* ap = As + (wave * 32 + s * 16 + ln) * 32 + hi * 8;
      af[s].q[0] = *(const uint4*)(ap);
      af[s].q[1] = *(const uint4*)(ap + 16);
    }
#pragma unroll
    for (int t = 0; t < 4; ++t) {
      union { v16bf v; uint4 q[2]; } bfr;
      const __bf16* bp = Bs + (t * 16 + ln) * 32 + hi * 8;
      bfr.q[0] = *(const uint4*)(bp);
      bfr.q[1] = *(const uint4*)(bp + 16);
#pragma unroll
      for (int s = 0; s < 2; ++s)
        acc[s][t] = __builtin_amdgcn_wmma_f32_16x16x32_bf16(
            false, af[s].v, false, bfr.v, (short)0, acc[s][t], false, false);
    }
    __syncthreads();
  }

#pragma unroll
  for (int t = 0; t < 4; ++t) {
    int col = n0 + t * 16 + ln;
    float bv = bias[col];
#pragma unroll
    for (int s = 0; s < 2; ++s) {
#pragma unroll
      for (int r = 0; r < 8; ++r) {
        int row = m0 + wave * 32 + s * 16 + r + 8 * hi;
        float v = acc[s][t][r] + bv;
        if (OUT_F32)
          ((float*)Cout)[(size_t)row * 512 + col] = v;
        else
          ((__bf16*)Cout)[(size_t)row * 512 + col] = (__bf16)v;
      }
    }
  }
}

// --------------------------- flash attention -------------------------------
// Grid: B*H*(L/128) = 1024 blocks x 256 threads. Wave w owns a 16-query tile.
// 64-key blocks: 8 score WMMAs + one softmax reduction pass + 8 O WMMAs.
__global__ void __launch_bounds__(256)
attention_wmma(const __bf16* __restrict__ Q, const __bf16* __restrict__ Km,
               const __bf16* __restrict__ Vm, __bf16* __restrict__ O) {
  __shared__ __bf16 Ks[64 * 64];       // 8 KB  K tile [key][d]
  __shared__ __bf16 Vt[64 * 64];       // 8 KB  V tile transposed [d][key]
  __shared__ __bf16 Ps[8 * 16 * 64];   // 16 KB per-wave P tiles [m][key]

  const int tid  = threadIdx.x;
  const int lane = tid & 31;
  const int wave = tid >> 5;
  const int ln   = lane & 15;
  const int hi   = lane >> 4;

  const int blk  = blockIdx.x;
  const int qblk = blk & 7;
  const int h    = (blk >> 3) & 7;
  const int b    = blk >> 6;

  const size_t base = ((size_t)b * NL) * NHID + (size_t)h * NHD;
  const int q0 = qblk * 128 + wave * 16;

  union frag { v16bf v; uint4 q[2]; };

  // Q fragments for the two d-steps (d 0..31, 32..63), loaded once
  frag aQ[2];
  {
    const __bf16* qp = Q + base + (size_t)(q0 + ln) * NHID;
#pragma unroll
    for (int s = 0; s < 2; ++s) {
      aQ[s].q[0] = *(const uint4*)(qp + s * 32 + hi * 8);
      aQ[s].q[1] = *(const uint4*)(qp + s * 32 + hi * 8 + 16);
    }
  }

  v8f acc[4] = {};
  float rmax[8], rsum[8];
#pragma unroll
  for (int r = 0; r < 8; ++r) { rmax[r] = -1e30f; rsum[r] = 0.0f; }

  __bf16* Pw = Ps + wave * 16 * 64;

  for (int kb = 0; kb < NL; kb += 64) {
    // cooperative stage: K as-is (2x16B/thread), V transposed (16 b16 stores)
    {
      int kk = tid >> 2;            // 0..63 key in tile
      int dc = (tid & 3) * 16;      // 0,16,32,48
      const uint4* ksrc = (const uint4*)(Km + base + (size_t)(kb + kk) * NHID + dc);
      uint4* kdst = (uint4*)(Ks + kk * 64 + dc);
      kdst[0] = ksrc[0];
      kdst[1] = ksrc[1];
      union { uint4 u[2]; __bf16 e[16]; } vv;
      const uint4* vsrc = (const uint4*)(Vm + base + (size_t)(kb + kk) * NHID + dc);
      vv.u[0] = vsrc[0];
      vv.u[1] = vsrc[1];
#pragma unroll
      for (int j = 0; j < 16; ++j) Vt[(dc + j) * 64 + kk] = vv.e[j];
    }
    __syncthreads();

    // scores: four 16-key tiles, d (=64) in two WMMA steps each
    v8f s[4] = {};
#pragma unroll
    for (int st = 0; st < 4; ++st) {
#pragma unroll
      for (int kd = 0; kd < 2; ++kd) {
        frag bK;
        const __bf16* kp = Ks + (st * 16 + ln) * 64 + kd * 32 + hi * 8;
        bK.q[0] = *(const uint4*)(kp);
        bK.q[1] = *(const uint4*)(kp + 16);
        s[st] = __builtin_amdgcn_wmma_f32_16x16x32_bf16(
            false, aQ[kd].v, false, bK.v, (short)0, s[st], false, false);
      }
    }

    // online softmax per output row: one butterfly pass covers 64 keys
    float p[4][8], scal[8];
#pragma unroll
    for (int r = 0; r < 8; ++r) {
      float v0 = s[0][r] * 0.125f, v1 = s[1][r] * 0.125f;   // 1/sqrt(64)
      float v2 = s[2][r] * 0.125f, v3 = s[3][r] * 0.125f;
      float mx = fmaxf(fmaxf(v0, v1), fmaxf(v2, v3));
#pragma unroll
      for (int off = 1; off <= 8; off <<= 1)
        mx = fmaxf(mx, __shfl_xor(mx, off, 32));
      float nm = fmaxf(rmax[r], mx);
      scal[r] = __expf(rmax[r] - nm);
      rmax[r] = nm;
      p[0][r] = __expf(v0 - nm);
      p[1][r] = __expf(v1 - nm);
      p[2][r] = __expf(v2 - nm);
      p[3][r] = __expf(v3 - nm);
      float ps = (p[0][r] + p[1][r]) + (p[2][r] + p[3][r]);
#pragma unroll
      for (int off = 1; off <= 8; off <<= 1)
        ps += __shfl_xor(ps, off, 32);
      rsum[r] = rsum[r] * scal[r] + ps;
    }
#pragma unroll
    for (int t = 0; t < 4; ++t)
#pragma unroll
      for (int r = 0; r < 8; ++r) acc[t][r] *= scal[r];

    // P tile -> per-wave LDS (score layout -> row-major [m][key])
#pragma unroll
    for (int st = 0; st < 4; ++st)
#pragma unroll
      for (int r = 0; r < 8; ++r)
        Pw[(r + 8 * hi) * 64 + st * 16 + ln] = (__bf16)p[st][r];

    // P as A-fragments (2 k-steps over the 64 keys); same-wave DS in-order
    frag aP[2];
#pragma unroll
    for (int s2 = 0; s2 < 2; ++s2) {
      const __bf16* pp = Pw + ln * 64 + s2 * 32 + hi * 8;
      aP[s2].q[0] = *(const uint4*)(pp);
      aP[s2].q[1] = *(const uint4*)(pp + 16);
    }
#pragma unroll
    for (int t = 0; t < 4; ++t) {
#pragma unroll
      for (int s2 = 0; s2 < 2; ++s2) {
        frag bV;
        const __bf16* vp = Vt + (t * 16 + ln) * 64 + s2 * 32 + hi * 8;
        bV.q[0] = *(const uint4*)(vp);
        bV.q[1] = *(const uint4*)(vp + 16);
        acc[t] = __builtin_amdgcn_wmma_f32_16x16x32_bf16(
            false, aP[s2].v, false, bV.v, (short)0, acc[t], false, false);
      }
    }
    __syncthreads();
  }

  // normalize by row sums, store bf16 [B,L,H,HD] (flat [M,512])
#pragma unroll
  for (int t = 0; t < 4; ++t) {
#pragma unroll
    for (int r = 0; r < 8; ++r) {
      int m = q0 + r + 8 * hi;
      int d = t * 16 + ln;
      O[base + (size_t)m * NHID + d] = (__bf16)(acc[t][r] / rsum[r]);
    }
  }
}

// ------------------------------ BatchNorm ----------------------------------
__global__ void zero_f32_kernel(float* p, int n) {
  int i = blockIdx.x * blockDim.x + threadIdx.x;
  if (i < n) p[i] = 0.0f;
}

__global__ void __launch_bounds__(256)
bn_partial_kernel(const float* __restrict__ X, float* __restrict__ sum,
                  float* __restrict__ sq) {
  int tid = threadIdx.x;
  int r0  = blockIdx.x * 128;
  float s0 = 0.f, s1 = 0.f, q0 = 0.f, q1 = 0.f;
  for (int r = 0; r < 128; ++r) {
    float a = X[(size_t)(r0 + r) * 512 + tid];
    float b = X[(size_t)(r0 + r) * 512 + tid + 256];
    s0 += a; q0 += a * a;
    s1 += b; q1 += b * b;
  }
  atomicAdd(&sum[tid], s0);       atomicAdd(&sq[tid], q0);
  atomicAdd(&sum[tid + 256], s1); atomicAdd(&sq[tid + 256], q1);
}

__global__ void bn_relu_kernel(const float* __restrict__ X,
                               const float* __restrict__ sum,
                               const float* __restrict__ sq,
                               const float* __restrict__ gamma,
                               const float* __restrict__ beta,
                               float* __restrict__ Y, int n) {
  int i = blockIdx.x * blockDim.x + threadIdx.x;
  if (i >= n) return;
  int c = i & 511;
  const float inv = 1.0f / (float)NM;
  float mean = sum[c] * inv;
  float var  = sq[c] * inv - mean * mean;
  float v = (X[i] - mean) * rsqrtf(var + EPSN) * gamma[c] + beta[c];
  Y[i] = fmaxf(v, 0.0f);
}

// ------------------------------ launcher -----------------------------------
extern "C" void kernel_launch(void* const* d_in, const int* in_sizes, int n_in,
                              void* d_out, int out_size, void* d_ws,
                              size_t ws_size, hipStream_t stream) {
  const float* value = (const float*)d_in[0];
  const float* query = (const float*)d_in[1];
  const float* Wq    = (const float*)d_in[2];
  const float* bq    = (const float*)d_in[3];
  const float* Wk    = (const float*)d_in[4];
  const float* bk    = (const float*)d_in[5];
  const float* Wv    = (const float*)d_in[6];
  const float* bv    = (const float*)d_in[7];
  const float* Wo    = (const float*)d_in[8];
  const float* bo    = (const float*)d_in[9];
  const float* gamma = (const float*)d_in[10];
  const float* beta  = (const float*)d_in[11];

  // workspace carve-out (~131 MB total)
  char* ws = (char*)d_ws;
  size_t off = 0;
  auto alloc = [&](size_t bytes) -> void* {
    void* p = ws + off;
    off += (bytes + 255) & ~(size_t)255;
    return p;
  };
  __bf16* qbf  = (__bf16*)alloc((size_t)NM * DIN * 2);
  __bf16* vbf  = (__bf16*)alloc((size_t)NM * DIN * 2);
  __bf16* wqb  = (__bf16*)alloc((size_t)NHID * DIN * 2);
  __bf16* wkb  = (__bf16*)alloc((size_t)NHID * DIN * 2);
  __bf16* wvb  = (__bf16*)alloc((size_t)NHID * DIN * 2);
  __bf16* wob  = (__bf16*)alloc((size_t)NHID * NHID * 2);
  __bf16* Qp   = (__bf16*)alloc((size_t)NM * NHID * 2);
  __bf16* Kp   = (__bf16*)alloc((size_t)NM * NHID * 2);
  __bf16* Vp   = (__bf16*)alloc((size_t)NM * NHID * 2);
  __bf16* attn = (__bf16*)alloc((size_t)NM * NHID * 2);
  float*  xo   = (float*)alloc((size_t)NM * NHID * 4);
  float*  ssum = (float*)alloc(1024 * 4);
  float*  ssq  = ssum + 512;

  // 1) precision conversion
  const int nact = NM * DIN;
  f32_to_bf16_kernel<<<(nact + 255) / 256, 256, 0, stream>>>(query, qbf, nact);
  f32_to_bf16_kernel<<<(nact + 255) / 256, 256, 0, stream>>>(value, vbf, nact);
  const int nw = NHID * DIN;
  f32_to_bf16_kernel<<<(nw + 255) / 256, 256, 0, stream>>>(Wq, wqb, nw);
  f32_to_bf16_kernel<<<(nw + 255) / 256, 256, 0, stream>>>(Wk, wkb, nw);
  f32_to_bf16_kernel<<<(nw + 255) / 256, 256, 0, stream>>>(Wv, wvb, nw);
  f32_to_bf16_kernel<<<(nw + 255) / 256, 256, 0, stream>>>(Wo, wob, nw);

  // 2) Q/K/V projections (WMMA, bf16 out), 256x64 block tiles
  dim3 gg(NM / 256, NHID / 64);
  gemm_bf16_wmma<false><<<gg, 256, 0, stream>>>(qbf, wqb, bq, Qp);
  gemm_bf16_wmma<false><<<gg, 256, 0, stream>>>(vbf, wkb, bk, Kp);
  gemm_bf16_wmma<false><<<gg, 256, 0, stream>>>(vbf, wvb, bv, Vp);

  // 3) flash attention (WMMA)
  attention_wmma<<<NB * NH * (NL / 128), 256, 0, stream>>>(Qp, Kp, Vp, attn);

  // 4) output projection (WMMA, fp32 out for BN stats)
  gemm_bf16_wmma<true><<<gg, 256, 0, stream>>>(attn, wob, bo, xo);

  // 5) BatchNorm (training stats over B*L per channel) + ReLU
  zero_f32_kernel<<<4, 256, 0, stream>>>(ssum, 1024);
  bn_partial_kernel<<<NM / 128, 256, 0, stream>>>(xo, ssum, ssq);
  const int ntot = NM * NHID;
  bn_relu_kernel<<<(ntot + 255) / 256, 256, 0, stream>>>(
      xo, ssum, ssq, gamma, beta, (float*)d_out, ntot);
}
